// ProtNgramGCN_4123168604624
// MI455X (gfx1250) — matrix operand order repaired and checked by reading, not compiled, not run on hardware
//
#include <hip/hip_runtime.h>
#include <hip/hip_bf16.h>
#include <math.h>

// ---------------------------------------------------------------------------
// ProtNgramGCN on MI455X (gfx1250): bf16 WMMA GEMMs + L2-resident scatter-add
// ---------------------------------------------------------------------------

typedef __bf16 bf16;
typedef __attribute__((ext_vector_type(16))) __bf16 v16bf;
typedef __attribute__((ext_vector_type(8)))  float v8f;

#define A_STRIDE 40  // LDS row stride in halfs (32 data + 8 pad, keeps 16B align)

// ---------------------------------------------------------------------------
// Weight prepack: W[K x N] fp32 row-major -> per-(ntile,ktile) WMMA B fragments.
// B fragment layout (16x16x32 bf16): lane n holds column N = n&15,
// half h holds K = kt*32 + ((lane&16)?16:0) + h.
// Packed element: dst[((ntOff+ntl)*Ktiles + kt)*512 + lane*16 + h]
// ---------------------------------------------------------------------------
__global__ void pack_weight_kernel(const float* __restrict__ W, int K, int N,
                                   bf16* __restrict__ dst, int ntOff, int Ktiles,
                                   int total) {
  int idx = blockIdx.x * blockDim.x + threadIdx.x;
  if (idx >= total) return;
  int h    = idx & 15;
  int lane = (idx >> 4) & 31;
  int kt   = (idx >> 9) % Ktiles;
  int ntl  = (idx >> 9) / Ktiles;
  int n = ntl * 16 + (lane & 15);
  int k = kt * 32 + ((lane & 16) ? 16 : 0) + h;
  float v = (k < K && n < N) ? W[(size_t)k * N + n] : 0.0f;
  dst[((size_t)(ntOff + ntl) * Ktiles + kt) * 512 + lane * 16 + h] = (bf16)v;
}

// ---------------------------------------------------------------------------
// Positional encoding + fp32 -> bf16 activation conversion.
// ---------------------------------------------------------------------------
__global__ void pe_bf16_kernel(const float* __restrict__ x, const float* __restrict__ pe,
                               bf16* __restrict__ xb, int total) {
  int idx = blockIdx.x * blockDim.x + threadIdx.x;
  if (idx >= total) return;
  int j = idx % 192;
  xb[idx] = (bf16)(x[idx] + pe[j]);  // pe rows 0..2 contiguous: pe[(j>>6)*64+(j&63)] == pe[j]
}

// ---------------------------------------------------------------------------
// Tiled bf16 WMMA GEMM:  C[M x Nfull] = A[M x K] @ Wpacked
// Block = 256 threads (8 waves). Block tile = 128 rows x 64 cols.
// Wave w owns rows [blk*128 + w*16, +16) and 4 accumulator tiles along N.
// Double-buffered LDS A tile (one barrier per K-step); KTILES templated so the
// K-loop fully unrolls and B-fragment loads pipeline under the WMMAs.
// A fragment layout: lane l: M=l&15, halves 0..15 -> K=kb+0..15,
// kb=(l&16)?16:0 (matches packed B fragment K order).
// ---------------------------------------------------------------------------
template <int KTILES>
__global__ __launch_bounds__(256) void gemm_bf16_wmma(
    const bf16* __restrict__ A, int K,
    const bf16* __restrict__ Wp,
    float* __restrict__ C, int M, int Nfull) {
  __shared__ __align__(16) bf16 As[2][128 * A_STRIDE];
  const int tid    = threadIdx.x;
  const int lane   = tid & 31;
  const int wave   = tid >> 5;
  const int rowBlk = blockIdx.x * 128;
  const int ntBase = blockIdx.y * 4;

  v8f acc[4] = {};

  const int kbase = (lane & 16) ? 16 : 0;
  const int mloc  = (wave * 16 + (lane & 15)) * A_STRIDE;

  // cooperative A-tile loader: 2 threads per row, 16 halfs (32B) each
  const int ldr = tid >> 1;
  const int ldo = (tid & 1) * 16;
  const int gm  = rowBlk + ldr;
  const v16bf* wp = (const v16bf*)Wp;

  uint4 p0, p1;  // global->LDS staging registers (next tile prefetch)
  auto loadTile = [&](int kt) {
    p0 = make_uint4(0, 0, 0, 0);
    p1 = make_uint4(0, 0, 0, 0);
    if (gm < M) {
      const uint4* src = (const uint4*)(A + (size_t)gm * K + kt * 32 + ldo);
      p0 = src[0];
      p1 = src[1];
    }
  };
  auto storeTile = [&](int buf) {
    uint4* dst = (uint4*)&As[buf][ldr * A_STRIDE + ldo];
    dst[0] = p0;
    dst[1] = p1;
  };

  loadTile(0);
  storeTile(0);

#pragma unroll
  for (int kt = 0; kt < KTILES; ++kt) {
    __syncthreads();  // As[kt&1] writes from previous iteration visible

    // issue next tile's global loads early: drained only by storeTile below,
    // so their latency hides under this iteration's WMMAs
    if (kt + 1 < KTILES) loadTile(kt + 1);

    // B fragments (L2-resident packed weights), coalesced 32B/lane
    v16bf bfrag[4];
#pragma unroll
    for (int t = 0; t < 4; ++t)
      bfrag[t] = wp[((size_t)(ntBase + t) * KTILES + kt) * 32 + lane];

    // A fragment from LDS
    union { uint4 u[2]; v16bf v; } af;
    const bf16* mr = &As[kt & 1][mloc];
    af.u[0] = *(const uint4*)&mr[kbase];
    af.u[1] = *(const uint4*)&mr[kbase + 8];

#pragma unroll
    for (int t = 0; t < 4; ++t)
      acc[t] = __builtin_amdgcn_wmma_f32_16x16x32_bf16(
          false, af.v, false, bfrag[t], (short)0, acc[t], false, false);

    if (kt + 1 < KTILES) storeTile((kt + 1) & 1);
  }

  // C/D layout: VGPR r -> lanes 0-15: M=r, N=lane; lanes 16-31: M=r+8, N=lane-16
  const int mrow0 = rowBlk + wave * 16 + ((lane & 16) ? 8 : 0);
  const int ncol  = lane & 15;
#pragma unroll
  for (int t = 0; t < 4; ++t) {
    const int n = (ntBase + t) * 16 + ncol;
#pragma unroll
    for (int r = 0; r < 8; ++r) {
      const int m = mrow0 + r;
      if (m < M) C[(size_t)m * Nfull + n] = acc[t][r];
    }
  }
}

// ---------------------------------------------------------------------------
// Fuse main + skip feature streams (prop is linear: prop(a)+prop(b)=prop(a+b)):
//   Zin  = Z[:,   0:128] + Z[:,256:384]
//   Zout = Z[:, 128:256] + Z[:,256:384]
// ---------------------------------------------------------------------------
__global__ void fuse_kernel(const float* __restrict__ Z, int Nz,
                            float* __restrict__ Zin, float* __restrict__ Zout,
                            int total) {
  int idx = blockIdx.x * blockDim.x + threadIdx.x;
  if (idx >= total) return;
  int node = idx >> 7, f = idx & 127;
  const float* zr = Z + (size_t)node * Nz;
  float s = zr[256 + f];
  Zin[idx]  = zr[f] + s;
  Zout[idx] = zr[128 + f] + s;
}

// ---------------------------------------------------------------------------
// Edge scatter: acc[dst] += w * Zs[src]. One wave per edge (32 lanes x 4 floats
// = full 128-float row, fully coalesced gather). Atomics stay in 192MB L2.
// ---------------------------------------------------------------------------
__global__ __launch_bounds__(256) void scatter_kernel(
    const int* __restrict__ ei, const float* __restrict__ ew,
    const float* __restrict__ Zs, float* __restrict__ acc, int E) {
  long long tid = (long long)blockIdx.x * blockDim.x + threadIdx.x;
  int e = (int)(tid >> 5);
  if (e >= E) return;
  int c = ((int)tid & 31) * 4;
  int src = ei[e];
  int dst = ei[E + e];
  float wgt = ew[e];
  const float4 v = *(const float4*)(Zs + (size_t)src * 128 + c);
  float* a = acc + (size_t)dst * 128 + c;
  unsafeAtomicAdd(a + 0, wgt * v.x);
  unsafeAtomicAdd(a + 1, wgt * v.y);
  unsafeAtomicAdd(a + 2, wgt * v.z);
  unsafeAtomicAdd(a + 3, wgt * v.w);
}

// ---------------------------------------------------------------------------
// Combine: h = tanh(cin*(accI + bmi + bsi) + cout*(accO + bmo + bso) + res)
// Layer1 residual comes from Z columns 384:512 (x_pe @ res1_w) + res1_b.
// Writes fp32 (next residual / final norm) and bf16 (next GEMM input).
// ---------------------------------------------------------------------------
__global__ void combine_kernel(
    const float* __restrict__ accI, const float* __restrict__ accO,
    const float* __restrict__ bmi, const float* __restrict__ bmo,
    const float* __restrict__ bsi, const float* __restrict__ bso,
    const float* __restrict__ cI,  const float* __restrict__ cO,
    const float* __restrict__ res, const float* __restrict__ Zres,
    const float* __restrict__ res1b,
    float* __restrict__ hOut, bf16* __restrict__ hb, int total) {
  int idx = blockIdx.x * blockDim.x + threadIdx.x;
  if (idx >= total) return;
  int node = idx >> 7, f = idx & 127;
  float vin  = accI[idx] + bmi[f] + bsi[f];
  float vout = accO[idx] + bmo[f] + bso[f];
  float r = Zres ? (Zres[(size_t)node * 512 + 384 + f] + res1b[f]) : res[idx];
  float h = tanhf(cI[node] * vin + cO[node] * vout + r);
  hOut[idx] = h;
  hb[idx] = (bf16)h;
}

// ---------------------------------------------------------------------------
// log_softmax over 20 classes + L2-normalized embedding output.
// d_out = [log_softmax (M x 20) | h3/(||h3||+eps) (M x 128)]
// ---------------------------------------------------------------------------
__global__ void out_kernel(const float* __restrict__ zl, const float* __restrict__ db,
                           const float* __restrict__ h3, float* __restrict__ out,
                           int M) {
  int n = blockIdx.x * blockDim.x + threadIdx.x;
  if (n >= M) return;
  float l[20];
  float mx = -1e30f;
  const float* zr = zl + (size_t)n * 64;
#pragma unroll
  for (int c = 0; c < 20; ++c) { l[c] = zr[c] + db[c]; mx = fmaxf(mx, l[c]); }
  float s = 0.0f;
#pragma unroll
  for (int c = 0; c < 20; ++c) s += __expf(l[c] - mx);
  float lse = mx + __logf(s);
  float* o1 = out + (size_t)n * 20;
#pragma unroll
  for (int c = 0; c < 20; ++c) o1[c] = l[c] - lse;

  const float* hr = h3 + (size_t)n * 128;
  float ss = 0.0f;
#pragma unroll 8
  for (int j = 0; j < 128; ++j) { float v = hr[j]; ss += v * v; }
  float inv = 1.0f / (sqrtf(ss) + 1e-12f);
  float* o2 = out + (size_t)M * 20 + (size_t)n * 128;
#pragma unroll 8
  for (int j = 0; j < 128; ++j) o2[j] = hr[j] * inv;
}

// ---------------------------------------------------------------------------
// Host launcher
// ---------------------------------------------------------------------------
struct LayerP {
  const float *Wmi, *Wmo, *Ws, *bmi, *bmo, *bsi, *bso, *cin, *cout;
};

extern "C" void kernel_launch(void* const* d_in, const int* in_sizes, int n_in,
                              void* d_out, int out_size, void* d_ws, size_t ws_size,
                              hipStream_t stream) {
  (void)out_size; (void)ws_size;
  const int M = in_sizes[0] / 192;   // 50000
  const int E = in_sizes[1] / 2;     // 800000

  const float* x      = (const float*)d_in[0];
  const int*   ei_in  = (const int*)d_in[1];
  const float* ew_in  = (const float*)d_in[2];
  const int*   ei_out = (const int*)d_in[3];
  const float* ew_out = (const float*)d_in[4];

  LayerP L1, L2, L3;
  const float *pe, *res1_w, *res1_b, *dec_w, *dec_b;

  if (n_in >= 37) {
    auto fill = [&](LayerP& L, int i0) {
      L.Wmi = (const float*)d_in[i0 + 0]; L.Wmo = (const float*)d_in[i0 + 1];
      L.Ws  = (const float*)d_in[i0 + 2];
      L.bmi = (const float*)d_in[i0 + 3]; L.bmo = (const float*)d_in[i0 + 4];
      L.bsi = (const float*)d_in[i0 + 5]; L.bso = (const float*)d_in[i0 + 6];
      L.cin = (const float*)d_in[i0 + 7]; L.cout = (const float*)d_in[i0 + 8];
    };
    fill(L1, 5); fill(L2, 14); fill(L3, 23);
    pe     = (const float*)d_in[32];
    res1_w = (const float*)d_in[33];
    res1_b = (const float*)d_in[34];
    dec_w  = (const float*)d_in[35];
    dec_b  = (const float*)d_in[36];
  } else {
    // layer tuples passed as single concatenated flat blobs
    auto fill = [&](LayerP& L, const float* b, int fin, int fout) {
      size_t o = 0;
      L.Wmi = b + o; o += (size_t)fin * fout;
      L.Wmo = b + o; o += (size_t)fin * fout;
      L.Ws  = b + o; o += (size_t)fin * fout;
      L.bmi = b + o; o += fout; L.bmo = b + o; o += fout;
      L.bsi = b + o; o += fout; L.bso = b + o; o += fout;
      L.cin = b + o; o += M;    L.cout = b + o;
    };
    fill(L1, (const float*)d_in[5], 192, 128);
    fill(L2, (const float*)d_in[6], 128, 128);
    fill(L3, (const float*)d_in[7], 128, 128);
    pe     = (const float*)d_in[8];
    res1_w = (const float*)d_in[9];
    res1_b = (const float*)d_in[10];
    dec_w  = (const float*)d_in[11];
    dec_b  = (const float*)d_in[12];
  }

  // ---- workspace carve ----
  char* wsp = (char*)d_ws;
  auto carve = [&](size_t bytes) {
    char* p = wsp;
    wsp += (bytes + 255) & ~(size_t)255;
    return (void*)p;
  };
  bf16*  wp1  = (bf16*) carve((size_t)32 * 6 * 512 * 2);  // layer1: Nt=32, Kt=6
  bf16*  wp2  = (bf16*) carve((size_t)24 * 4 * 512 * 2);  // layer2: Nt=24, Kt=4
  bf16*  wp3  = (bf16*) carve((size_t)24 * 4 * 512 * 2);
  bf16*  wpd  = (bf16*) carve((size_t)4  * 4 * 512 * 2);  // decoder padded to N=64
  bf16*  xb   = (bf16*) carve((size_t)M * 192 * 2);
  bf16*  hb   = (bf16*) carve((size_t)M * 128 * 2);
  float* Z    = (float*)carve((size_t)M * 512 * 4);
  float* Zin  = (float*)carve((size_t)M * 128 * 4);
  float* Zout = (float*)carve((size_t)M * 128 * 4);
  float* accI = (float*)carve((size_t)M * 128 * 4);
  float* accO = (float*)carve((size_t)M * 128 * 4);
  float* hA   = (float*)carve((size_t)M * 128 * 4);
  float* hB   = (float*)carve((size_t)M * 128 * 4);
  float* zlog = (float*)carve((size_t)M * 64 * 4);

  // ---- pack all weights (tiny, recomputed every call: deterministic) ----
  auto packW = [&](const float* W, int K, int N, bf16* dst, int ntOff, int Kt, int ntCnt) {
    int total = ntCnt * Kt * 512;
    pack_weight_kernel<<<(total + 255) / 256, 256, 0, stream>>>(W, K, N, dst, ntOff, Kt, total);
  };
  packW(L1.Wmi, 192, 128, wp1,  0, 6, 8);
  packW(L1.Wmo, 192, 128, wp1,  8, 6, 8);
  packW(L1.Ws , 192, 128, wp1, 16, 6, 8);
  packW(res1_w, 192, 128, wp1, 24, 6, 8);
  packW(L2.Wmi, 128, 128, wp2,  0, 4, 8);
  packW(L2.Wmo, 128, 128, wp2,  8, 4, 8);
  packW(L2.Ws , 128, 128, wp2, 16, 4, 8);
  packW(L3.Wmi, 128, 128, wp3,  0, 4, 8);
  packW(L3.Wmo, 128, 128, wp3,  8, 4, 8);
  packW(L3.Ws , 128, 128, wp3, 16, 4, 8);
  packW(dec_w,  128,  20, wpd,  0, 4, 4);

  // ---- positional encoding + bf16 conversion ----
  {
    int total = M * 192;
    pe_bf16_kernel<<<(total + 255) / 256, 256, 0, stream>>>(x, pe, xb, total);
  }

  const int t128   = M * 128;
  const int gM     = (M + 127) / 128;
  const long long et = (long long)E * 32;
  const int gE     = (int)((et + 255) / 256);

  auto launchGemm = [&](const bf16* Ain, int K, const bf16* wp, int Kt,
                        float* Cout, int Nz) {
    dim3 g(gM, Nz / 64);
    if (Kt == 6)
      gemm_bf16_wmma<6><<<g, 256, 0, stream>>>(Ain, K, wp, Cout, M, Nz);
    else
      gemm_bf16_wmma<4><<<g, 256, 0, stream>>>(Ain, K, wp, Cout, M, Nz);
  };

  auto runLayer = [&](const bf16* in, int K, const bf16* wp, int Kt, int Nz,
                      const LayerP& L, const float* res, bool useZres, float* hOut) {
    launchGemm(in, K, wp, Kt, Z, Nz);
    fuse_kernel<<<(t128 + 255) / 256, 256, 0, stream>>>(Z, Nz, Zin, Zout, t128);
    hipMemsetAsync(accI, 0, (size_t)t128 * 4, stream);
    hipMemsetAsync(accO, 0, (size_t)t128 * 4, stream);
    scatter_kernel<<<gE, 256, 0, stream>>>(ei_in,  ew_in,  Zin,  accI, E);
    scatter_kernel<<<gE, 256, 0, stream>>>(ei_out, ew_out, Zout, accO, E);
    combine_kernel<<<(t128 + 255) / 256, 256, 0, stream>>>(
        accI, accO, L.bmi, L.bmo, L.bsi, L.bso, L.cin, L.cout,
        res, useZres ? Z : (const float*)nullptr, res1_b, hOut, hb, t128);
  };

  runLayer(xb, 192, wp1, 6, 512, L1, nullptr, true,  hA);  // h1 -> hA, hb
  runLayer(hb, 128, wp2, 4, 384, L2, hA,      false, hB);  // h2 -> hB, hb
  runLayer(hb, 128, wp3, 4, 384, L3, hB,      false, hA);  // h3 -> hA, hb

  // ---- decoder + outputs ----
  launchGemm(hb, 128, wpd, 4, zlog, 64);
  out_kernel<<<(M + 255) / 256, 256, 0, stream>>>(zlog, dec_b, hA, (float*)d_out, M);
}
